// PairformerLayer_21766894256700
// MI455X (gfx1250) — matrix operand order repaired and checked by reading, not compiled, not run on hardware
//
#include <hip/hip_runtime.h>
#include <hip/hip_bf16.h>

typedef unsigned short u16;
typedef unsigned int   u32;
typedef long long      i64;

typedef __attribute__((ext_vector_type(16))) __bf16 bf16x16;
typedef __attribute__((ext_vector_type(8)))  float  f32x8;

union FragU { u32 u[8]; bf16x16 v; };

enum { EP_STORE = 0, EP_ADD = 1, EP_ADD_T = 2, EP_SIGMOID = 3, EP_ADD_GATED = 4,
       EP_GLU_SIG_MASK = 5, EP_GLU_SILU = 6 };

__device__ __forceinline__ u16 f2bf(float f) {
  u32 x = __float_as_uint(f);
  return (u16)((x + 0x7FFFu + ((x >> 16) & 1u)) >> 16);   // RNE fp32->bf16
}
__device__ __forceinline__ float bf2f(u16 h) { return __uint_as_float(((u32)h) << 16); }
__device__ __forceinline__ float sigm(float x) { return 1.0f / (1.0f + __expf(-x)); }

__device__ __forceinline__ f32x8 wmma_bf(bf16x16 a, bf16x16 b, f32x8 c) {
  return __builtin_amdgcn_wmma_f32_16x16x32_bf16(false, a, false, b, (short)0, c, false, false);
}

// A fragment: 16x32 bf16 tile; row-major source, leading dim ld (elements).
// Layout: lanes0-15 row m, k 0..15; lanes16-31 row m, k 16..31 (matches frag_b).
__device__ __forceinline__ bf16x16 frag_a(const u16* base, int row0, int ld) {
  int lane = threadIdx.x & 31;
  const u32* p = (const u32*)(base + (i64)(row0 + (lane & 15)) * ld);
  int kq = (lane >> 4) << 3;
  FragU f;
#pragma unroll
  for (int g = 0; g < 4; ++g) f.u[g] = p[kq + g];
#pragma unroll
  for (int g = 0; g < 4; ++g) f.u[4 + g] = p[kq + 4 + g];
  return f.v;
}

// B fragment: 32x16 bf16 tile; source is B^T row-major [n][k], leading dim ld.
// Contiguous k order used here (not ISA table order): lane n holds k 0..15, lane n+16 holds k 16..31.
__device__ __forceinline__ bf16x16 frag_b(const u16* base, int col0, int ld) {
  int lane = threadIdx.x & 31;
  const u32* p = (const u32*)(base + (i64)(col0 + (lane & 15)) * ld + ((lane >> 4) << 4));
  FragU f;
#pragma unroll
  for (int g = 0; g < 8; ++g) f.u[g] = p[g];
  return f.v;
}

// ---------------------------------------------------------------------------
// Generic bf16 WMMA GEMM: C[M,NC] (+)= A[M,K] * B^T[NC,K], batched, fused epilogues.
// Block 256 thr = 8 waves (2x4), wave computes 32x32 via 2x2 16x16 WMMA tiles.
// ---------------------------------------------------------------------------
__global__ __launch_bounds__(256)
void gemm_bf16_kernel(const u16* __restrict__ A, const u16* __restrict__ B,
                      int M, int NC, int K,
                      i64 sA, i64 sB, i64 sC,
                      float* __restrict__ C, int ldc,
                      int ep, int tflag,
                      const float* __restrict__ gate,
                      const float* __restrict__ maskp,
                      u16* __restrict__ Obf, i64 sObf)
{
  const int lane = threadIdx.x & 31;
  const int wave = threadIdx.x >> 5;
  const int bm = blockIdx.y * 64 + (wave >> 2) * 32;
  const int bn = blockIdx.x * 128 + (wave & 3) * 32;
  if (bm >= M || bn >= NC) return;
  const u16* Ab = A + (i64)blockIdx.z * sA;
  const u16* Bb = B + (i64)blockIdx.z * sB;

  f32x8 acc[2][2];
#pragma unroll
  for (int i = 0; i < 2; ++i)
#pragma unroll
    for (int j = 0; j < 2; ++j)
#pragma unroll
      for (int e = 0; e < 8; ++e) acc[i][j][e] = 0.0f;

  for (int k0 = 0; k0 < K; k0 += 32) {
    bf16x16 a0 = frag_a(Ab + k0, bm,      K);
    bf16x16 a1 = frag_a(Ab + k0, bm + 16, K);
    bf16x16 b0 = frag_b(Bb + k0, bn,      K);
    bf16x16 b1 = frag_b(Bb + k0, bn + 16, K);
    acc[0][0] = wmma_bf(a0, b0, acc[0][0]);
    acc[0][1] = wmma_bf(a0, b1, acc[0][1]);
    acc[1][0] = wmma_bf(a1, b0, acc[1][0]);
    acc[1][1] = wmma_bf(a1, b1, acc[1][1]);
  }

  if (ep <= EP_ADD_GATED) {
    float* Cb = C + (i64)blockIdx.z * sC;
#pragma unroll
    for (int mi = 0; mi < 2; ++mi)
#pragma unroll
      for (int ni = 0; ni < 2; ++ni) {
        int col = bn + ni * 16 + (lane & 15);
#pragma unroll
        for (int g = 0; g < 8; ++g) {
          int row = bm + mi * 16 + g + ((lane >> 4) << 3);
          float v = acc[mi][ni][g];
          i64 o = (i64)row * ldc + col;
          if (ep == EP_STORE)        Cb[o] = v;
          else if (ep == EP_ADD)     Cb[o] += v;
          else if (ep == EP_ADD_T) { // transposed residual add (attention "ending")
            i64 dr = (i64)(((row & 255) << 8) | (row >> 8));
            Cb[dr * ldc + col] += v;
          }
          else if (ep == EP_SIGMOID) Cb[o] = sigm(v);
          else                       Cb[o] += gate[o] * v;  // EP_ADD_GATED
        }
      }
  } else {
    // GLU epilogues: n-tile0 = "gate half", n-tile1 = "value half" (16-col interleaved weights)
    int outcol = (bn >> 1) + (lane & 15);
#pragma unroll
    for (int mi = 0; mi < 2; ++mi)
#pragma unroll
      for (int g = 0; g < 8; ++g) {
        int row = bm + mi * 16 + g + ((lane >> 4) << 3);
        float a0 = acc[mi][0][g], a1 = acc[mi][1][g];
        if (ep == EP_GLU_SIG_MASK) {
          // tri-mul a/b: sigmoid(x@g)*(x@p)*pair_mask -> channel-major bf16 (opt transpose)
          float mv = maskp[row >> 8] * maskp[row & 255];
          i64 dr = tflag ? (i64)(((row & 255) << 8) | (row >> 8)) : (i64)row;
          Obf[(i64)outcol * sObf + dr] = f2bf(sigm(a0) * a1 * mv);
        } else {
          // transitions: silu(x@w1)*(x@w2) -> row-major bf16
          Obf[(i64)row * sObf + outcol] = f2bf(a0 * sigm(a0) * a1);
        }
      }
  }
}

// ---------------------------------------------------------------------------
// Fused attention: per block = one (head, query-slab). Q/K/V + bf16 scores in LDS.
// scores = QK^T*scale + bias + (mask-1)*1e9 ; softmax ; O = A*V.  All GEMMs WMMA.
// ---------------------------------------------------------------------------
#define SROW 264   // bf16 score row stride (pad vs 256 to break bank pattern)
#define ATTN_LDS ((256 * SROW + 3 * 256 * 32) * 2)

__global__ __launch_bounds__(256)
void attn_kernel(const u16* __restrict__ Q, const u16* __restrict__ Km,
                 const u16* __restrict__ V, const float* __restrict__ bias,
                 const float* __restrict__ mask, float* __restrict__ O,
                 int biasDiv, float scale)
{
  extern __shared__ char smem[];
  u16* Sb = (u16*)smem;                 // [256][SROW] bf16 scores
  u16* Qs = Sb + 256 * SROW;            // [256][32]
  u16* Ks = Qs + 256 * 32;              // [256][32]
  u16* Vs = Ks + 256 * 32;              // V^T: [32][256]
  const int tid = threadIdx.x;
  const int outer = blockIdx.x;
  const u16* Qg = Q  + (i64)outer * 8192;
  const u16* Kg = Km + (i64)outer * 8192;
  const u16* Vg = V  + (i64)outer * 8192;
  const float* bg = bias + (i64)(outer / biasDiv) * 65536;

  { // stage Q,K (as dwords) and V transposed
    const u32* q32 = (const u32*)Qg; u32* qs = (u32*)Qs;
    const u32* k32 = (const u32*)Kg; u32* ks = (u32*)Ks;
    for (int i = tid; i < 4096; i += 256) { qs[i] = q32[i]; ks[i] = k32[i]; }
    for (int i = tid; i < 8192; i += 256) Vs[(i & 31) * 256 + (i >> 5)] = Vg[i];
  }
  __syncthreads();
  const int lane = tid & 31, wave = tid >> 5;

  // Phase 1: S = Q K^T (d=32 -> single WMMA per 16x16 tile), fused scale+bias+mask
  for (int t = wave; t < 256; t += 8) {
    int tm = (t >> 4) << 4, tn = (t & 15) << 4;
    f32x8 acc;
#pragma unroll
    for (int e = 0; e < 8; ++e) acc[e] = 0.0f;
    acc = wmma_bf(frag_a(Qs, tm, 32), frag_b(Ks, tn, 32), acc);
    int col = tn + (lane & 15);
    float mt = (mask[col] - 1.0f) * 1e9f;
#pragma unroll
    for (int g = 0; g < 8; ++g) {
      int row = tm + g + ((lane >> 4) << 3);
      Sb[row * SROW + col] = f2bf(acc[g] * scale + bg[row * 256 + col] + mt);
    }
  }
  __syncthreads();

  // Phase 2: softmax, one row per thread, in place (bf16)
  {
    u16* sr = Sb + tid * SROW;
    float mx = -1e30f;
    for (int k = 0; k < 256; ++k) mx = fmaxf(mx, bf2f(sr[k]));
    float sum = 0.0f;
    for (int k = 0; k < 256; ++k) { float e = __expf(bf2f(sr[k]) - mx); sum += e; sr[k] = f2bf(e); }
    float inv = 1.0f / sum;
    for (int k = 0; k < 256; ++k) sr[k] = f2bf(bf2f(sr[k]) * inv);
  }
  __syncthreads();

  // Phase 3: O = A * V  (M=256, N=32, K=256)
  float* Og = O + (i64)outer * 8192;
  for (int t = wave; t < 32; t += 8) {
    int tm = (t >> 1) << 4, tn = (t & 1) << 4;
    f32x8 acc;
#pragma unroll
    for (int e = 0; e < 8; ++e) acc[e] = 0.0f;
    for (int k0 = 0; k0 < 256; k0 += 32)
      acc = wmma_bf(frag_a(Sb + k0, tm, SROW), frag_b(Vs + k0, tn, 256), acc);
    int col = tn + (lane & 15);
#pragma unroll
    for (int g = 0; g < 8; ++g) {
      int row = tm + g + ((lane >> 4) << 3);
      Og[row * 32 + col] = acc[g];
    }
  }
}

// ---------------------------------------------------------------------------
// LayerNorm: one wave per row, bf16 output (row-major). Optional (i,j)<->(j,i).
// ---------------------------------------------------------------------------
__global__ __launch_bounds__(256)
void ln_kernel(const float* __restrict__ X, const float* __restrict__ w,
               const float* __restrict__ b, u16* __restrict__ Y,
               int R, int C, int transpose)
{
  int row = blockIdx.x * 8 + (threadIdx.x >> 5);
  int lane = threadIdx.x & 31;
  if (row >= R) return;
  i64 src = transpose ? (i64)(((row & 255) << 8) | (row >> 8)) * C : (i64)row * C;
  float s = 0.f, s2 = 0.f;
  for (int c = lane; c < C; c += 32) { float v = X[src + c]; s += v; s2 += v * v; }
#pragma unroll
  for (int o = 16; o > 0; o >>= 1) { s += __shfl_xor(s, o, 32); s2 += __shfl_xor(s2, o, 32); }
  float m = s / C;
  float inv = rsqrtf(fmaxf(s2 / C - m * m, 0.f) + 1e-5f);
  for (int c = lane; c < C; c += 32)
    Y[(i64)row * C + c] = f2bf((X[src + c] - m) * inv * w[c] + b[c]);
}

// LayerNorm over channel-major input T[C][R] (triangle-einsum output)
__global__ __launch_bounds__(256)
void ln_cm_kernel(const float* __restrict__ T, const float* __restrict__ w,
                  const float* __restrict__ b, u16* __restrict__ Y, int R, int C)
{
  int row = blockIdx.x * 8 + (threadIdx.x >> 5);
  int lane = threadIdx.x & 31;
  if (row >= R) return;
  float s = 0.f, s2 = 0.f;
  for (int c = lane; c < C; c += 32) { float v = T[(i64)c * R + row]; s += v; s2 += v * v; }
#pragma unroll
  for (int o = 16; o > 0; o >>= 1) { s += __shfl_xor(s, o, 32); s2 += __shfl_xor(s2, o, 32); }
  float m = s / C;
  float inv = rsqrtf(fmaxf(s2 / C - m * m, 0.f) + 1e-5f);
  for (int c = lane; c < C; c += 32)
    Y[(i64)row * C + c] = f2bf((T[(i64)c * R + row] - m) * inv * w[c] + b[c]);
}

// ---------------------------------------------------------------------------
// Weight conversion / packing / repack helpers
// ---------------------------------------------------------------------------
__global__ __launch_bounds__(256)
void fill0_kernel(u16* __restrict__ p, i64 n) {
  i64 i = (i64)blockIdx.x * 256 + threadIdx.x;
  if (i < n) p[i] = 0;
}

// W fp32 [K,ncols] -> Wt bf16 [coloff+n][k]  (transposed to [n][k] for frag_b)
__global__ __launch_bounds__(256)
void convw_kernel(const float* __restrict__ W, u16* __restrict__ Wt,
                  int K, int ncols, int coloff)
{
  i64 idx = (i64)blockIdx.x * 256 + threadIdx.x;
  if (idx >= (i64)ncols * K) return;
  int n = (int)(idx / K), k = (int)(idx % K);
  Wt[(i64)(coloff + n) * K + k] = f2bf(W[(i64)k * ncols + n]);
}

// GLU packing: dest rows 32t..+15 = W1 cols 16t..; rows +16..+31 = W2 cols 16t..
__global__ __launch_bounds__(256)
void convw_glu_kernel(const float* __restrict__ W1, const float* __restrict__ W2,
                      u16* __restrict__ Wt, int K, int H)
{
  i64 idx = (i64)blockIdx.x * 256 + threadIdx.x;
  if (idx >= (i64)2 * H * K) return;
  int drow = (int)(idx / K), k = (int)(idx % K);
  int t = drow >> 5, r = drow & 31;
  const float* W = (r < 16) ? W1 : W2;
  int n = t * 16 + (r & 15);
  Wt[idx] = f2bf(W[(i64)k * H + n]);
}

// Y fp32 [R,ld] (q|k|v|g|...) -> q/k/v bf16 [(h*R+r)*32+d], zero-padded d>=Dsrc
__global__ __launch_bounds__(256)
void repack_qkv_kernel(const float* __restrict__ Y, int ld, int H, int Dsrc, int R,
                       const float* __restrict__ bq,
                       u16* __restrict__ q, u16* __restrict__ k, u16* __restrict__ v)
{
  i64 idx = (i64)blockIdx.x * 256 + threadIdx.x;
  if (idx >= (i64)R * H * 32) return;
  int d = (int)(idx & 31);
  i64 t = idx >> 5;
  int h = (int)(t % H);
  i64 r = t / H;
  i64 dst = ((i64)h * R + r) * 32 + d;
  if (d < Dsrc) {
    i64 s = r * ld + h * Dsrc + d;
    float qv = Y[s];
    if (bq) qv += bq[h * Dsrc + d];
    q[dst] = f2bf(qv);
    k[dst] = f2bf(Y[s + H * Dsrc]);
    v[dst] = f2bf(Y[s + 2 * H * Dsrc]);
  } else { q[dst] = 0; k[dst] = 0; v[dst] = 0; }
}

__global__ __launch_bounds__(256)
void repack_bias_kernel(const float* __restrict__ Y, int ld, int coloff, int H, i64 R,
                        float* __restrict__ bias)
{
  i64 idx = (i64)blockIdx.x * 256 + threadIdx.x;
  if (idx >= R * H) return;
  int h = (int)(idx % H);
  i64 r = idx / H;
  bias[(i64)h * R + r] = Y[r * ld + coloff + h];
}

// A_att = sigmoid(gate) * O   -> bf16 [r][h*Dsrc+d]  (input to output projection)
__global__ __launch_bounds__(256)
void gate_attn_kernel(const float* __restrict__ Y, int ld, int goff, int H, int Dsrc,
                      i64 R, const float* __restrict__ O, u16* __restrict__ Aout)
{
  i64 idx = (i64)blockIdx.x * 256 + threadIdx.x;
  if (idx >= R * H * Dsrc) return;
  int d = (int)(idx % Dsrc);
  i64 t = idx / Dsrc;
  int h = (int)(t % H);
  i64 r = t / H;
  float g = sigm(Y[r * ld + goff + h * Dsrc + d]);
  float o = O[((i64)h * R + r) * 32 + d];
  Aout[r * (i64)(H * Dsrc) + h * Dsrc + d] = f2bf(g * o);
}

// ---------------------------------------------------------------------------
// Host orchestration
// ---------------------------------------------------------------------------
#define PRM(i) ((const float*)d_in[(i)])

extern "C" void kernel_launch(void* const* d_in, const int* in_sizes, int n_in,
                              void* d_out, int out_size, void* d_ws, size_t ws_size,
                              hipStream_t stream) {
  (void)in_sizes; (void)n_in; (void)out_size; (void)ws_size;
  const float* mask = (const float*)d_in[2];
  const i64 R2 = 65536;                      // N*N pair rows

  // ---- workspace carve ----
  char* wp = (char*)d_ws;
  auto carve = [&](size_t bytes) { char* p = wp; wp += (bytes + 255) & ~(size_t)255; return p; };
  u16*   XBF  = (u16*)carve((size_t)R2 * 128 * 2);
  u16*   SNBF = (u16*)carve((size_t)256 * 384 * 2);
  u16*   QB   = (u16*)carve((size_t)4 * R2 * 32 * 2);
  u16*   KB   = (u16*)carve((size_t)4 * R2 * 32 * 2);
  u16*   VB   = (u16*)carve((size_t)4 * R2 * 32 * 2);
  float* BIAS = (float*)carve((size_t)16 * R2 * 4);
  float* OB   = (float*)carve((size_t)4 * R2 * 32 * 4);
  u16*   AATT = (u16*)carve((size_t)R2 * 128 * 2);
  char*  BIG  = carve((size_t)R2 * 544 * 4);           // stage-local overlay region
  // overlays within BIG
  u16*   ABUF = (u16*)BIG;                              // tri-mul a  [128][65536]
  u16*   BBUF = (u16*)(BIG + (size_t)128 * R2 * 2);     // tri-mul b
  float* GATE = (float*)(BIG + (size_t)256 * R2 * 2);   // sigmoid(x@g)
  float* TBUF = (float*)(BIG + (size_t)256 * R2 * 2 + (size_t)R2 * 128 * 4);
  u16*   TLN  = (u16*)(BIG + (size_t)256 * R2 * 2 + (size_t)2 * R2 * 128 * 4);
  float* Y    = (float*)BIG;                            // qkvb projections [65536,544]
  u16*   UBUF = (u16*)BIG;                              // transition hidden bf16
  float* YS   = (float*)BIG;                            // pb qkvg [256,1536]
  float* YZ   = (float*)(BIG + (size_t)256 * 1536 * 4); // pb z-bias [65536,32]
  // weights (bf16, pre-transposed to [n][k])
  u16 *Wglu1[2], *Wglu2[2], *Wg[2], *Wout[2], *Wqkvb[2], *Wo[2];
  for (int t = 0; t < 2; ++t) {
    Wglu1[t] = (u16*)carve(256 * 128 * 2);
    Wglu2[t] = (u16*)carve(256 * 128 * 2);
    Wg[t]    = (u16*)carve(128 * 128 * 2);
    Wout[t]  = (u16*)carve(128 * 128 * 2);
  }
  for (int t = 0; t < 2; ++t) {
    Wqkvb[t] = (u16*)carve(544 * 128 * 2);
    Wo[t]    = (u16*)carve(128 * 128 * 2);
  }
  u16* Wqkvg   = (u16*)carve((size_t)1536 * 384 * 2);
  u16* Wz      = (u16*)carve(32 * 128 * 2);
  u16* Wopb    = (u16*)carve((size_t)384 * 384 * 2);
  u16* Wglu_tz = (u16*)carve((size_t)1024 * 128 * 2);
  u16* W3tz    = (u16*)carve((size_t)128 * 512 * 2);
  u16* Wglu_ts = (u16*)carve((size_t)3072 * 384 * 2);
  u16* W3ts    = (u16*)carve((size_t)384 * 1536 * 2);

  // ---- launch helpers ----
  auto convw = [&](const float* W, u16* Wt, int K, int ncols, int coloff) {
    i64 n = (i64)ncols * K;
    convw_kernel<<<dim3((unsigned)((n + 255) / 256)), dim3(256), 0, stream>>>(W, Wt, K, ncols, coloff);
  };
  auto convglu = [&](const float* W1, const float* W2, u16* Wt, int K, int H) {
    i64 n = (i64)2 * H * K;
    convw_glu_kernel<<<dim3((unsigned)((n + 255) / 256)), dim3(256), 0, stream>>>(W1, W2, Wt, K, H);
  };
  auto fill0 = [&](u16* p, i64 n) {
    fill0_kernel<<<dim3((unsigned)((n + 255) / 256)), dim3(256), 0, stream>>>(p, n);
  };
  auto gemm = [&](const u16* A, const u16* B, int M, int NC, int K, int batch,
                  i64 sA, i64 sB, i64 sC, float* C, int ldc, int ep, int tflag,
                  const float* gatep, u16* obf, i64 sobf) {
    dim3 grid((unsigned)((NC + 127) / 128), (unsigned)((M + 63) / 64), (unsigned)batch);
    gemm_bf16_kernel<<<grid, dim3(256), 0, stream>>>(A, B, M, NC, K, sA, sB, sC,
                                                     C, ldc, ep, tflag, gatep, mask, obf, sobf);
  };
  auto ln = [&](const float* X, const float* g, const float* b, u16* Yo, int R, int C, int tr) {
    ln_kernel<<<dim3((unsigned)((R + 7) / 8)), dim3(256), 0, stream>>>(X, g, b, Yo, R, C, tr);
  };

  // ---- convert all weights to bf16 (transposed / interleaved / padded) ----
  for (int t = 0; t < 2; ++t) {               // tmo, tmi
    int b = 3 + 10 * t;
    convglu(PRM(b + 2), PRM(b + 3), Wglu1[t], 128, 128);  // g1|p1
    convglu(PRM(b + 4), PRM(b + 5), Wglu2[t], 128, 128);  // g2|p2
    convw(PRM(b + 8), Wg[t], 128, 128, 0);
    convw(PRM(b + 9), Wout[t], 128, 128, 0);
  }
  for (int t = 0; t < 2; ++t) {               // tas, tae
    int b = 23 + 8 * t;
    fill0(Wqkvb[t], 544 * 128);
    convw(PRM(b + 2), Wqkvb[t], 128, 128, 0);    // wq
    convw(PRM(b + 3), Wqkvb[t], 128, 128, 128);  // wk
    convw(PRM(b + 4), Wqkvb[t], 128, 128, 256);  // wv
    convw(PRM(b + 6), Wqkvb[t], 128, 128, 384);  // wg
    convw(PRM(b + 5), Wqkvb[t], 128, 4, 512);    // wb (padded to 32 cols)
    convw(PRM(b + 7), Wo[t], 128, 128, 0);       // wo
  }
  convw(PRM(41), Wqkvg, 384, 384, 0);      // pb_wq
  convw(PRM(43), Wqkvg, 384, 384, 384);    // pb_wk
  convw(PRM(44), Wqkvg, 384, 384, 768);    // pb_wv
  convw(PRM(45), Wqkvg, 384, 384, 1152);   // pb_wg
  fill0(Wz, 32 * 128);
  convw(PRM(48), Wz, 128, 16, 0);          // pb_wz (padded to 32)
  convw(PRM(49), Wopb, 384, 384, 0);       // pb_wo
  convglu(PRM(52), PRM(53), Wglu_tz, 128, 512);
  convw(PRM(54), W3tz, 512, 128, 0);
  convglu(PRM(57), PRM(58), Wglu_ts, 384, 1536);
  convw(PRM(59), W3ts, 1536, 384, 0);

  // ---- outputs: s then z (residual chains live in d_out) ----
  float* s_out = (float*)d_out;
  float* z_out = s_out + 256 * 384;
  hipMemcpyAsync(s_out, d_in[0], (size_t)256 * 384 * 4, hipMemcpyDeviceToDevice, stream);
  hipMemcpyAsync(z_out, d_in[1], (size_t)R2 * 128 * 4, hipMemcpyDeviceToDevice, stream);

  // ---- triangle multiplications (t=0 outgoing, t=1 incoming) ----
  for (int t = 0; t < 2; ++t) {
    int b = 3 + 10 * t;
    ln(z_out, PRM(b + 0), PRM(b + 1), XBF, 65536, 128, 0);
    gemm(XBF, Wglu1[t], 65536, 256, 128, 1, 0, 0, 0, nullptr, 0,
         EP_GLU_SIG_MASK, t, nullptr, ABUF, R2);
    gemm(XBF, Wglu2[t], 65536, 256, 128, 1, 0, 0, 0, nullptr, 0,
         EP_GLU_SIG_MASK, t, nullptr, BBUF, R2);
    gemm(XBF, Wg[t], 65536, 128, 128, 1, 0, 0, 0, GATE, 128, EP_SIGMOID, 0, nullptr, nullptr, 0);
    // triangle einsum: 128 batched 256x256x256 GEMMs, channel-major operands
    gemm(ABUF, BBUF, 256, 256, 256, 128, R2, R2, R2, TBUF, 256, EP_STORE, 0, nullptr, nullptr, 0);
    ln_cm_kernel<<<dim3(8192), dim3(256), 0, stream>>>(TBUF, PRM(b + 6), PRM(b + 7), TLN, 65536, 128);
    gemm(TLN, Wout[t], 65536, 128, 128, 1, 0, 0, 0, z_out, 128, EP_ADD_GATED, 0, GATE, nullptr, 0);
  }

  // ---- triangle attentions (t=0 starting, t=1 ending/transposed) ----
  for (int t = 0; t < 2; ++t) {
    int b = 23 + 8 * t;
    ln(z_out, PRM(b + 0), PRM(b + 1), XBF, 65536, 128, t);
    gemm(XBF, Wqkvb[t], 65536, 544, 128, 1, 0, 0, 0, Y, 544, EP_STORE, 0, nullptr, nullptr, 0);
    { i64 n = (i64)R2 * 4 * 32;
      repack_qkv_kernel<<<dim3((unsigned)((n + 255) / 256)), dim3(256), 0, stream>>>(
          Y, 544, 4, 32, 65536, nullptr, QB, KB, VB); }
    { i64 n = (i64)R2 * 4;
      repack_bias_kernel<<<dim3((unsigned)((n + 255) / 256)), dim3(256), 0, stream>>>(
          Y, 544, 512, 4, R2, BIAS); }
    attn_kernel<<<dim3(1024), dim3(256), ATTN_LDS, stream>>>(
        QB, KB, VB, BIAS, mask, OB, 256, 0.17677669529663687f /*1/sqrt(32)*/);
    { i64 n = (i64)R2 * 128;
      gate_attn_kernel<<<dim3((unsigned)((n + 255) / 256)), dim3(256), 0, stream>>>(
          Y, 544, 384, 4, 32, R2, OB, AATT); }
    gemm(AATT, Wo[t], 65536, 128, 128, 1, 0, 0, 0, z_out, 128,
         t ? EP_ADD_T : EP_ADD, 0, nullptr, nullptr, 0);
  }

  // ---- pair transition (SwiGLU) ----
  ln(z_out, PRM(50), PRM(51), XBF, 65536, 128, 0);
  gemm(XBF, Wglu_tz, 65536, 1024, 128, 1, 0, 0, 0, nullptr, 0, EP_GLU_SILU, 0, nullptr, UBUF, 512);
  gemm(UBUF, W3tz, 65536, 128, 512, 1, 0, 0, 0, z_out, 128, EP_ADD, 0, nullptr, nullptr, 0);

  // ---- pair-bias attention on s ----
  ln(s_out, PRM(39), PRM(40), SNBF, 256, 384, 0);
  gemm(SNBF, Wqkvg, 256, 1536, 384, 1, 0, 0, 0, YS, 1536, EP_STORE, 0, nullptr, nullptr, 0);
  { i64 n = (i64)256 * 16 * 32;
    repack_qkv_kernel<<<dim3((unsigned)((n + 255) / 256)), dim3(256), 0, stream>>>(
        YS, 1536, 16, 24, 256, PRM(42), QB, KB, VB); }
  ln(z_out, PRM(46), PRM(47), XBF, 65536, 128, 0);
  gemm(XBF, Wz, 65536, 32, 128, 1, 0, 0, 0, YZ, 32, EP_STORE, 0, nullptr, nullptr, 0);
  { i64 n = (i64)R2 * 16;
    repack_bias_kernel<<<dim3((unsigned)((n + 255) / 256)), dim3(256), 0, stream>>>(
        YZ, 32, 0, 16, R2, BIAS); }
  attn_kernel<<<dim3(16), dim3(256), ATTN_LDS, stream>>>(
      QB, KB, VB, BIAS, mask, OB, 1, 0.2041241452319315f /*1/sqrt(24)*/);
  { i64 n = (i64)256 * 384;
    gate_attn_kernel<<<dim3((unsigned)((n + 255) / 256)), dim3(256), 0, stream>>>(
        YS, 1536, 1152, 16, 24, 256, OB, AATT); }
  gemm(AATT, Wopb, 256, 384, 384, 1, 0, 0, 0, s_out, 384, EP_ADD, 0, nullptr, nullptr, 0);

  // ---- sequence transition (SwiGLU) ----
  ln(s_out, PRM(55), PRM(56), SNBF, 256, 384, 0);
  gemm(SNBF, Wglu_ts, 256, 3072, 384, 1, 0, 0, 0, nullptr, 0, EP_GLU_SILU, 0, nullptr, UBUF, 1536);
  gemm(UBUF, W3ts, 256, 384, 1536, 1, 0, 0, 0, s_out, 384, EP_ADD, 0, nullptr, nullptr, 0);
}
